// AAMBlock_77927886619032
// MI455X (gfx1250) — matrix operand order repaired and checked by reading, not compiled, not run on hardware
//
#include <hip/hip_runtime.h>
#include <hip/hip_bf16.h>
#include <math.h>

// MI455X / gfx1250, wave32. Flash-attention style spatial attention block.
// B=4, C=256, Cqk=32, N=4096. f16 WMMA (16x16x32) with f32 accumulation.
// Kernel 2: wave-independent flash attention, 64 keys per chunk, base-2
// softmax (q pre-scaled by log2(e)), lane-local deferred sum reduction,
// ds_swizzle XOR butterflies for the per-chunk row-max.

typedef __attribute__((ext_vector_type(16))) _Float16 v16h;
typedef __attribute__((ext_vector_type(8)))  _Float16 v8h;
typedef __attribute__((ext_vector_type(8)))  float    v8f;

union HV16 { v16h v; v8h h[2]; };

#define B_   4
#define C_   256
#define CQK_ 32
#define N_   4096
#define LOG2E_ 1.4426950408889634f

// ds_swizzle_b32, group-of-32 mode: offset[14:10]=and_mask, [9:5]=or_mask,
// [4:0]=xor_mask. Full and-mask + xor -> butterfly exchange, no VALU setup.
#define SWZ_XOR(v, m) \
  __int_as_float(__builtin_amdgcn_ds_swizzle(__float_as_int(v), (0x1f << 10) | (m)))

// ------------------------------------------------------------------
// Kernel 1: q/k/v 1x1-conv projections, cast to f16 in workspace.
//   qh[b][m][cqk]  (cqk contiguous, pre-scaled by log2e) -> WMMA A friendly
//   kh[b][n][cqk]  (cqk contiguous)  -> WMMA B-matrix friendly (K dim = c)
//   vh[b][c][n]    (n contiguous)    -> WMMA B-matrix friendly (K dim = n)
// ------------------------------------------------------------------
__global__ __launch_bounds__(256) void proj_qkv_kernel(
    const float* __restrict__ x,
    const float* __restrict__ Wq, const float* __restrict__ bq,
    const float* __restrict__ Wk, const float* __restrict__ bk,
    const float* __restrict__ Wv, const float* __restrict__ bv,
    _Float16* __restrict__ qh, _Float16* __restrict__ kh,
    _Float16* __restrict__ vh)
{
  const int bn = blockIdx.x;
  const int b  = bn >> 12;
  const int n  = bn & (N_ - 1);
  const int t  = threadIdx.x;

  __shared__ __align__(16) float xcol[C_];
  xcol[t] = x[((size_t)(b * C_ + t)) * N_ + n];
  __syncthreads();

  const float4* xc4 = (const float4*)xcol;

  // v channel t
  {
    const float4* wr = (const float4*)(Wv + (size_t)t * C_);
    float acc = bv[t];
#pragma unroll 8
    for (int i = 0; i < C_ / 4; ++i) {
      float4 w = wr[i], xv = xc4[i];
      acc = fmaf(w.x, xv.x, acc); acc = fmaf(w.y, xv.y, acc);
      acc = fmaf(w.z, xv.z, acc); acc = fmaf(w.w, xv.w, acc);
    }
    vh[((size_t)(b * C_ + t)) * N_ + n] = (_Float16)acc;
  }
  // q (threads 0..31) and k (threads 32..63)
  if (t < 2 * CQK_) {
    const int    c    = t & (CQK_ - 1);
    const float* W    = (t < CQK_) ? Wq : Wk;
    const float* bias = (t < CQK_) ? bq : bk;
    const float4* wr = (const float4*)(W + (size_t)c * C_);
    float acc = bias[c];
#pragma unroll 8
    for (int i = 0; i < C_ / 4; ++i) {
      float4 w = wr[i], xv = xc4[i];
      acc = fmaf(w.x, xv.x, acc); acc = fmaf(w.y, xv.y, acc);
      acc = fmaf(w.z, xv.z, acc); acc = fmaf(w.w, xv.w, acc);
    }
    if (t < CQK_) {
      // fold log2(e) into q: lets the softmax use native base-2 v_exp_f32
      qh[((size_t)b * N_ + n) * CQK_ + c] = (_Float16)(acc * LOG2E_);
    } else {
      kh[((size_t)b * N_ + n) * CQK_ + c] = (_Float16)acc;
    }
  }
}

// ------------------------------------------------------------------
// Kernel 2: flash attention. 8 waves/block, each wave independent:
//   wave handles m-tile (blockIdx*4 + wave/2), channels [128*(wave&1), +128).
//   Per 64-key chunk: 4 energy WMMAs + base-2 softmax + 16 P·V WMMAs.
// Row max reduced per chunk (4 ds_swizzle steps); row sums stay lane-local
// and are reduced once at the end. No __syncthreads anywhere.
// ------------------------------------------------------------------
__global__ __launch_bounds__(256) void flash_attn_kernel(
    const _Float16* __restrict__ qh, const _Float16* __restrict__ kh,
    const _Float16* __restrict__ vh, float* __restrict__ o_ws)
{
  const int lane = threadIdx.x & 31;
  const int wave = threadIdx.x >> 5;
  const int lid  = lane & 15;
  const int hi   = lane >> 4;          // rows m = hi*8 + r in C/D layout

  const int mt = blockIdx.x * 4 + (wave >> 1);  // m-tile id, 0..1023
  const int b  = mt >> 8;
  const int m0 = (mt & 255) << 4;
  const int c0 = (wave & 1) * (C_ / 2);         // this wave's channel half

  __shared__ __align__(16) _Float16 p_lds_all[8][16][64];
  _Float16 (*p_lds)[64] = p_lds_all[wave];      // private per-wave slice

  // Q tile in WMMA A layout:
  // lane<16: row m0+lid, K chunks {0..7},{16..23}; lane>=16: {8..15},{24..31}
  HV16 qa;
  {
    const _Float16* qp = qh + ((size_t)b * N_ + m0 + lid) * CQK_ + hi * 8;
    qa.h[0] = *(const v8h*)(qp);
    qa.h[1] = *(const v8h*)(qp + 16);
  }

  float rmax[8], rsum[8];
#pragma unroll
  for (int r = 0; r < 8; ++r) { rmax[r] = -INFINITY; rsum[r] = 0.0f; }

  v8f acc[8];
  {
    v8f z = {};
#pragma unroll
    for (int ct = 0; ct < 8; ++ct) acc[ct] = z;
  }

  // K in B layout: column n = n0+16j+lid, K(c) = hi*8 + [0..7], 16 + hi*8 + [0..7]
  const _Float16* kbase = kh + (size_t)b * N_ * CQK_ + (size_t)lid * CQK_ + hi * 8;
  // V in B layout: column c = c0+16*ct+lid, K(n) = nbase + hi*8 + [0..7], nbase + 16 + hi*8 + [0..7]
  const _Float16* vbase = vh + ((size_t)(b * C_ + c0 + lid)) * N_ + hi * 8;

#pragma unroll 1
  for (int n0 = 0; n0 < N_; n0 += 64) {
    HV16 kb[4];
#pragma unroll
    for (int j = 0; j < 4; ++j) {
      const _Float16* kj = kbase + (size_t)(n0 + 16 * j) * CQK_;
      kb[j].h[0] = *(const v8h*)(kj);
      kb[j].h[1] = *(const v8h*)(kj + 16);
    }
    v8f zero = {};
    v8f e[4];
#pragma unroll
    for (int j = 0; j < 4; ++j)
      e[j] = __builtin_amdgcn_wmma_f32_16x16x32_f16(
          false, qa.v, false, kb[j].v, (short)0, zero, false, false);

    float scl[8];
#pragma unroll
    for (int r = 0; r < 8; ++r) {
      float a0 = e[0][r], a1 = e[1][r], a2 = e[2][r], a3 = e[3][r];
      float mx = fmaxf(fmaxf(a0, a1), fmaxf(a2, a3));
      mx = fmaxf(mx, SWZ_XOR(mx, 1));
      mx = fmaxf(mx, SWZ_XOR(mx, 2));
      mx = fmaxf(mx, SWZ_XOR(mx, 4));
      mx = fmaxf(mx, SWZ_XOR(mx, 8));
      float nm = fmaxf(rmax[r], mx);
      float sc = __builtin_amdgcn_exp2f(rmax[r] - nm);  // first iter: 2^-inf=0
      float p0 = __builtin_amdgcn_exp2f(a0 - nm);
      float p1 = __builtin_amdgcn_exp2f(a1 - nm);
      float p2 = __builtin_amdgcn_exp2f(a2 - nm);
      float p3 = __builtin_amdgcn_exp2f(a3 - nm);
      // lane-local partial sum; cross-lane reduction deferred to the end
      rsum[r] = rsum[r] * sc + ((p0 + p1) + (p2 + p3));
      rmax[r] = nm;
      scl[r]  = sc;
      const int row = hi * 8 + r;
      p_lds[row][lid]      = (_Float16)p0;
      p_lds[row][lid + 16] = (_Float16)p1;
      p_lds[row][lid + 32] = (_Float16)p2;
      p_lds[row][lid + 48] = (_Float16)p3;
    }

    // Same-wave DS ops complete in order; make the cross-lane transpose
    // visible and keep the compiler from reordering around it.
    asm volatile("s_wait_dscnt 0x0" ::: "memory");

    // P tiles in A layout from this wave's LDS slice (16B-aligned b128s).
    HV16 pa0, pa1;
    pa0.h[0] = *(const v8h*)&p_lds[lid][hi * 8];
    pa0.h[1] = *(const v8h*)&p_lds[lid][16 + hi * 8];
    pa1.h[0] = *(const v8h*)&p_lds[lid][32 + hi * 8];
    pa1.h[1] = *(const v8h*)&p_lds[lid][48 + hi * 8];

#pragma unroll
    for (int ct = 0; ct < 8; ++ct) {
      const _Float16* vp = vbase + (size_t)ct * 16 * N_ + n0;
      HV16 vb0, vb1;
      vb0.h[0] = *(const v8h*)(vp);
      vb0.h[1] = *(const v8h*)(vp + 16);
      vb1.h[0] = *(const v8h*)(vp + 32);
      vb1.h[1] = *(const v8h*)(vp + 48);
#pragma unroll
      for (int r = 0; r < 8; ++r) acc[ct][r] *= scl[r];
      acc[ct] = __builtin_amdgcn_wmma_f32_16x16x32_f16(
          false, pa0.v, false, vb0.v, (short)0, acc[ct], false, false);
      acc[ct] = __builtin_amdgcn_wmma_f32_16x16x32_f16(
          false, pa1.v, false, vb1.v, (short)0, acc[ct], false, false);
    }
  }

  // One-time cross-lane reduction of the deferred row sums.
  float inv[8];
#pragma unroll
  for (int r = 0; r < 8; ++r) {
    float s = rsum[r];
    s += SWZ_XOR(s, 1);
    s += SWZ_XOR(s, 2);
    s += SWZ_XOR(s, 4);
    s += SWZ_XOR(s, 8);
    inv[r] = 1.0f / s;
  }

#pragma unroll
  for (int ct = 0; ct < 8; ++ct) {
#pragma unroll
    for (int r = 0; r < 8; ++r) {
      const int m = m0 + hi * 8 + r;
      o_ws[((size_t)b * N_ + m) * C_ + c0 + ct * 16 + lid] = acc[ct][r] * inv[r];
    }
  }
}

// ------------------------------------------------------------------
// Kernel 3: out = Wo @ concat([attn_out, x]) + bo
// ------------------------------------------------------------------
__global__ __launch_bounds__(256) void out_proj_kernel(
    const float* __restrict__ x, const float* __restrict__ o_ws,
    const float* __restrict__ Wo, const float* __restrict__ bo,
    float* __restrict__ out)
{
  const int bn = blockIdx.x;
  const int b  = bn >> 12;
  const int m  = bn & (N_ - 1);
  const int t  = threadIdx.x;

  __shared__ __align__(16) float col[2 * C_];
  col[t]      = o_ws[((size_t)b * N_ + m) * C_ + t];      // coalesced
  col[C_ + t] = x[((size_t)(b * C_ + t)) * N_ + m];       // strided gather
  __syncthreads();

  const float4* c4 = (const float4*)col;
  const float4* wr = (const float4*)(Wo + (size_t)t * (2 * C_));
  float acc = bo[t];
#pragma unroll 8
  for (int i = 0; i < (2 * C_) / 4; ++i) {
    float4 w = wr[i], v = c4[i];
    acc = fmaf(w.x, v.x, acc); acc = fmaf(w.y, v.y, acc);
    acc = fmaf(w.z, v.z, acc); acc = fmaf(w.w, v.w, acc);
  }
  out[((size_t)(b * C_ + t)) * N_ + m] = acc;
}

// ------------------------------------------------------------------
extern "C" void kernel_launch(void* const* d_in, const int* in_sizes, int n_in,
                              void* d_out, int out_size, void* d_ws, size_t ws_size,
                              hipStream_t stream) {
  (void)in_sizes; (void)n_in; (void)out_size; (void)ws_size;
  const float* x  = (const float*)d_in[0];
  const float* Wq = (const float*)d_in[1];
  const float* bq = (const float*)d_in[2];
  const float* Wk = (const float*)d_in[3];
  const float* bk = (const float*)d_in[4];
  const float* Wv = (const float*)d_in[5];
  const float* bv = (const float*)d_in[6];
  const float* Wo = (const float*)d_in[7];
  const float* bo = (const float*)d_in[8];
  float* out = (float*)d_out;

  // Workspace carve-up (~26 MB total, fits in L2):
  _Float16* qh = (_Float16*)d_ws;                       // 1 MB
  _Float16* kh = qh + (size_t)B_ * N_ * CQK_;           // 1 MB
  _Float16* vh = kh + (size_t)B_ * N_ * CQK_;           // 8 MB
  float*  o_ws = (float*)(vh + (size_t)B_ * N_ * C_);   // 16 MB

  proj_qkv_kernel<<<B_ * N_, 256, 0, stream>>>(x, Wq, bq, Wk, bk, Wv, bv,
                                               qh, kh, vh);
  // 1024 m-tiles x 2 waves each = 2048 waves = 256 blocks of 8 waves.
  flash_attn_kernel<<<256, 256, 0, stream>>>(qh, kh, vh, o_ws);
  out_proj_kernel<<<B_ * N_, 256, 0, stream>>>(x, o_ws, Wo, bo, out);
}
